// EFFT_35407710388860
// MI455X (gfx1250) — matrix operand order repaired and checked by reading, not compiled, not run on hardware
//
#include <hip/hip_runtime.h>
#include <math.h>

// ---------------- problem constants (match reference) ----------------
constexpr int Bn    = 8;
constexpr int Nn    = 8192;
constexpr int Sn    = 512;    // NPOINT
constexpr int Kn    = 32;     // NSAMPLE
constexpr int Dn    = 128;
constexpr int NH    = 8;
constexpr int HDn   = 16;
constexpr int DFFn  = 256;
constexpr int NL    = 4;
constexpr int DOUT  = 256;
constexpr int MT    = Bn * Sn * Kn;   // 131072 token rows
constexpr float EPSc = 1e-5f;

// ---------------- WMMA types / fragment loaders ----------------
typedef __attribute__((ext_vector_type(16))) _Float16     v16h;
typedef __attribute__((ext_vector_type(8)))  float        v8f;
typedef __attribute__((ext_vector_type(4)))  unsigned int u32x4;

union H16 { v16h v; _Float16 h[16]; unsigned int u[8]; u32x4 q[2]; };

// A fragment: 16x32 f16 tile, element (m,k) at p[m*ldk + k]
// lane layout: m = lane&15; lanes<16 hold k 0..15, lanes>=16 hold k 16..31
__device__ __forceinline__ v16h load_a_frag(const _Float16* p, int ldk, int lane) {
  H16 r;
  int m  = lane & 15;
  int kb = (lane >> 4) * 16;
  const _Float16* row = p + (size_t)m * ldk;
  r.q[0] = *(const u32x4*)(row + kb);        // k = kb .. kb+7
  r.q[1] = *(const u32x4*)(row + kb + 8);    // k = kb+8 .. kb+15
  return r.v;
}

// A fragment with only k<16 valid (upper K half zero) — for Q@K^T with HD=16
__device__ __forceinline__ v16h load_a_frag_k16(const _Float16* p, int ldk, int lane) {
  H16 r;
  int m  = lane & 15;
  const _Float16* row = p + (size_t)m * ldk;
  u32x4 z = {0u, 0u, 0u, 0u};
  r.q[0] = (lane < 16) ? *(const u32x4*)(row) : z;
  r.q[1] = (lane < 16) ? *(const u32x4*)(row + 8) : z;
  return r.v;
}

// B fragment: 32x16 f16 tile, element (k,n) at p[n*ldk + k]  (i.e. weight row-major [N,K])
// lane layout: n = lane&15; lanes<16 hold k 0..15, lanes>=16 hold k 16..31 (2 per VGPR)
__device__ __forceinline__ v16h load_b_frag(const _Float16* p, int ldk, int lane) {
  H16 r;
  int n  = lane & 15;
  int kb = (lane >> 4) * 16;
  const _Float16* row = p + (size_t)n * ldk + kb;
  r.q[0] = *(const u32x4*)(row);
  r.q[1] = *(const u32x4*)(row + 8);
  return r.v;
}

// B fragment with only k<16 valid (lanes>=16 all zero) — K^T tile for attention scores
__device__ __forceinline__ v16h load_bT_frag_k16(const _Float16* p, int ldk, int lane) {
  H16 r;
  u32x4 z = {0u, 0u, 0u, 0u};
  int n = lane & 15;
  if (lane < 16) {
    const _Float16* row = p + (size_t)n * ldk;
    r.q[0] = *(const u32x4*)(row);
    r.q[1] = *(const u32x4*)(row + 8);
  } else {
    r.q[0] = z;
    r.q[1] = z;
  }
  return r.v;
}

// ---------------- fp32 -> f16 weight conversion ----------------
__global__ void cvt_f16_kernel(const float* __restrict__ src, _Float16* __restrict__ dst, int n) {
  int i = blockIdx.x * blockDim.x + threadIdx.x;
  if (i < n) dst[i] = (_Float16)src[i];
}

// ---------------- FPS: one block per batch ----------------
__global__ void fps_kernel(const float* __restrict__ xyz, float* __restrict__ newXyz) {
  __shared__ float rv[256];
  __shared__ int   ri[256];
  __shared__ int   sIdx[Sn];
  __shared__ float cen[3];
  __shared__ int   curIdx;
  const int b = blockIdx.x;
  const int t = threadIdx.x;
  const float* px = xyz + (size_t)b * 3 * Nn;
  const float* py = px + Nn;
  const float* pz = py + Nn;

  float d[32];
#pragma unroll
  for (int j = 0; j < 32; ++j) d[j] = 1e10f;
  if (t == 0) curIdx = 0;
  __syncthreads();

  for (int it = 0; it < Sn; ++it) {
    int far = curIdx;
    if (t == 0) {
      sIdx[it] = far;
      cen[0] = px[far]; cen[1] = py[far]; cen[2] = pz[far];
    }
    __syncthreads();
    float cx = cen[0], cy = cen[1], cz = cen[2];
    float best = -1.0f; int bi = 0;
#pragma unroll 4
    for (int j = 0; j < 32; ++j) {
      int i = t + 256 * j;
      float dx = px[i] - cx, dy = py[i] - cy, dz = pz[i] - cz;
      float dd = dx * dx + dy * dy + dz * dz;
      float nd = fminf(d[j], dd);
      d[j] = nd;
      if (nd > best) { best = nd; bi = i; }
    }
    rv[t] = best; ri[t] = bi;
    __syncthreads();
    for (int s = 128; s >= 1; s >>= 1) {
      if (t < s) {
        float ov = rv[t + s]; int oi = ri[t + s];
        if (ov > rv[t] || (ov == rv[t] && oi < ri[t])) { rv[t] = ov; ri[t] = oi; }
      }
      __syncthreads();
    }
    if (t == 0) curIdx = ri[0];
    __syncthreads();
  }
  for (int s = t; s < Sn; s += 256) {
    int idx = sIdx[s];
    float* o = newXyz + ((size_t)b * Sn + s) * 3;
    o[0] = px[idx]; o[1] = py[idx]; o[2] = pz[idx];
  }
}

// ---------------- kNN: one block per (b,s) ----------------
__global__ void knn_kernel(const float* __restrict__ xyz, const float* __restrict__ newXyz,
                           int* __restrict__ gidx) {
  __shared__ float dist[Nn];      // 32 KB
  __shared__ float rv[256];
  __shared__ int   ri[256];
  const int bs = blockIdx.x;       // b*Sn + s
  const int b  = bs / Sn;
  const int t  = threadIdx.x;
  const float* px = xyz + (size_t)b * 3 * Nn;
  const float* py = px + Nn;
  const float* pz = py + Nn;
  const float* c  = newXyz + (size_t)bs * 3;
  float cx = c[0], cy = c[1], cz = c[2];

  for (int i = t; i < Nn; i += 256) {
    float dx = px[i] - cx, dy = py[i] - cy, dz = pz[i] - cz;
    dist[i] = dx * dx + dy * dy + dz * dz;
  }
  __syncthreads();

  for (int kk = 0; kk < Kn; ++kk) {
    float best = 1e30f; int bi = 0;
#pragma unroll 4
    for (int j = 0; j < Nn / 256; ++j) {
      int i = t + 256 * j;
      float v = dist[i];
      if (v < best) { best = v; bi = i; }
    }
    rv[t] = best; ri[t] = bi;
    __syncthreads();
    for (int s = 128; s >= 1; s >>= 1) {
      if (t < s) {
        float ov = rv[t + s]; int oi = ri[t + s];
        if (ov < rv[t] || (ov == rv[t] && oi < ri[t])) { rv[t] = ov; ri[t] = oi; }
      }
      __syncthreads();
    }
    if (t == 0) {
      gidx[(size_t)bs * Kn + kk] = ri[0];
      dist[ri[0]] = 1e30f;
    }
    __syncthreads();
  }
}

// ---------------- gather + positional-encoding MLP -> X (fp32) and Xh (f16) ----------------
__global__ void build_x_kernel(const float* __restrict__ xyz, const float* __restrict__ feat,
                               const int* __restrict__ gidx,
                               const float* __restrict__ pe_w1, const float* __restrict__ pe_b1,
                               const float* __restrict__ bn_g, const float* __restrict__ bn_b,
                               const float* __restrict__ bn_m, const float* __restrict__ bn_v,
                               const float* __restrict__ pe_w2, const float* __restrict__ pe_b2,
                               float* __restrict__ X, _Float16* __restrict__ Xh) {
  __shared__ float h[64];
  __shared__ float g3[3];
  const int bs = blockIdx.x;           // b*Sn + s
  const int b  = bs / Sn;
  const int t  = threadIdx.x;          // 0..127 == output dim

  for (int k = 0; k < Kn; ++k) {
    __syncthreads();
    int gi = gidx[(size_t)bs * Kn + k];
    if (t < 3) g3[t] = xyz[(size_t)b * 3 * Nn + (size_t)t * Nn + gi];
    __syncthreads();
    if (t < 64) {
      float hv = g3[0] * pe_w1[t * 3 + 0] + g3[1] * pe_w1[t * 3 + 1] + g3[2] * pe_w1[t * 3 + 2] + pe_b1[t];
      hv = (hv - bn_m[t]) * rsqrtf(bn_v[t] + EPSc) * bn_g[t] + bn_b[t];
      h[t] = fmaxf(hv, 0.0f);
    }
    __syncthreads();
    float pe = pe_b2[t];
#pragma unroll 8
    for (int j = 0; j < 64; ++j) pe += h[j] * pe_w2[t * 64 + j];
    float gf = feat[(size_t)b * Dn * Nn + (size_t)t * Nn + gi];
    float xv = gf + pe;
    size_t row = (size_t)bs * Kn + k;
    X[row * Dn + t]  = xv;
    Xh[row * Dn + t] = (_Float16)xv;
  }
}

// ---------------- WMMA GEMM with async-to-LDS weight staging ----------------
// C[M,N] = A[M,KD] @ W[N,KD]^T + bias.
// Block = 8 waves = 128 rows (8 M-tiles) x 64 cols (4 N-tiles).
// Weight tile (64 rows x KD, contiguous) is staged into LDS via
// GLOBAL_LOAD_ASYNC_TO_LDS_B128 (ASYNCcnt), then B fragments come from LDS.
template <int KD, bool RELU>
__global__ void wmma_gemm(const _Float16* __restrict__ A, const _Float16* __restrict__ W,
                          const float* __restrict__ bias, int Ntiles,
                          _Float16* __restrict__ outH, float* __restrict__ outF) {
  __shared__ _Float16 sW[64 * KD];   // 16 KB (KD=128) / 32 KB (KD=256)
  const int lane = threadIdx.x & 31;
  const int wave = threadIdx.x >> 5;
  const int nblocks = Ntiles >> 2;           // 4 N-tiles per block
  const int mblk = blockIdx.x / nblocks;
  const int nblk = blockIdx.x % nblocks;

  // ---- async stage of the contiguous 64xKD weight tile into LDS ----
  {
    unsigned ldsbase = (unsigned)(unsigned long long)
        (__attribute__((address_space(3))) _Float16*)sW;
    const char* gsrc = (const char*)(W + (size_t)nblk * 64 * KD);
    const int chunks = (64 * KD * 2) / 16;   // 16B per lane per op
    for (int i = threadIdx.x; i < chunks; i += 256) {
      unsigned lo = ldsbase + (unsigned)(i * 16);
      unsigned long long ga = (unsigned long long)(gsrc + (size_t)i * 16);
      asm volatile("global_load_async_to_lds_b128 %0, %1, off"
                   :: "v"(lo), "v"(ga) : "memory");
    }
    asm volatile("s_wait_asynccnt 0x0" ::: "memory");
  }
  __syncthreads();

  // ---- register-blocked mainloop: 1 A frag amortized over 4 N-tiles ----
  const int mtile = mblk * 8 + wave;
  const _Float16* Ab = A + (size_t)mtile * 16 * KD;

  v8f acc0 = {}, acc1 = {}, acc2 = {}, acc3 = {};
#pragma unroll
  for (int kk = 0; kk < KD; kk += 32) {
    v16h a = load_a_frag(Ab + kk, KD, lane);
    v16h b0 = load_b_frag(sW + 0  * 16 * KD + kk, KD, lane);
    v16h b1 = load_b_frag(sW + 1  * 16 * KD + kk, KD, lane);
    v16h b2 = load_b_frag(sW + 2  * 16 * KD + kk, KD, lane);
    v16h b3 = load_b_frag(sW + 3  * 16 * KD + kk, KD, lane);
    acc0 = __builtin_amdgcn_wmma_f32_16x16x32_f16(false, a, false, b0, (short)0, acc0, false, false);
    acc1 = __builtin_amdgcn_wmma_f32_16x16x32_f16(false, a, false, b1, (short)0, acc1, false, false);
    acc2 = __builtin_amdgcn_wmma_f32_16x16x32_f16(false, a, false, b2, (short)0, acc2, false, false);
    acc3 = __builtin_amdgcn_wmma_f32_16x16x32_f16(false, a, false, b3, (short)0, acc3, false, false);
  }

  // ---- epilogue ----
  const int N = Ntiles * 16;
  v8f accs[4] = {acc0, acc1, acc2, acc3};
#pragma unroll
  for (int t = 0; t < 4; ++t) {
    int n = nblk * 64 + t * 16 + (lane & 15);
    float bv = bias[n];
#pragma unroll
    for (int r = 0; r < 8; ++r) {
      int m = mtile * 16 + ((lane >> 4) * 8) + r;
      float v = accs[t][r] + bv;
      if (RELU) v = fmaxf(v, 0.0f);
      size_t idx = (size_t)m * N + n;
      if (outF) outF[idx] = v;
      if (outH) outH[idx] = (_Float16)v;
    }
  }
}

// ---------------- fused attention: one block per sequence (32 tokens), one wave per head ----------------
__global__ void attn_kernel(const _Float16* __restrict__ qkv, _Float16* __restrict__ obuf) {
  __shared__ float    sS[NH][32][32];   // 32 KB scores
  __shared__ _Float16 sP[NH][32][32];   // 16 KB probs
  const int lane = threadIdx.x & 31;
  const int head = threadIdx.x >> 5;
  const size_t R = (size_t)blockIdx.x * 32;   // token row base
  const _Float16* q  = qkv + R * 384 + head * HDn;
  const _Float16* kp = q + Dn;
  const _Float16* vp = q + 2 * Dn;

  // S = (Q @ K^T) * rsqrt(HD);  K dim = 16, padded to 32 with zeros
#pragma unroll
  for (int mt = 0; mt < 2; ++mt) {
    v16h a = load_a_frag_k16(q + (size_t)mt * 16 * 384, 384, lane);
#pragma unroll
    for (int nt = 0; nt < 2; ++nt) {
      v16h b = load_bT_frag_k16(kp + (size_t)nt * 16 * 384, 384, lane);
      v8f c = {};
      c = __builtin_amdgcn_wmma_f32_16x16x32_f16(false, a, false, b, (short)0, c, false, false);
      int n = nt * 16 + (lane & 15);
#pragma unroll
      for (int r = 0; r < 8; ++r) {
        int m = mt * 16 + (lane >> 4) * 8 + r;
        sS[head][m][n] = c[r] * 0.25f;   // 1/sqrt(16)
      }
    }
  }
  __syncthreads();

  // per-row softmax: lane == row
  {
    float mx = -1e30f;
#pragma unroll 8
    for (int j = 0; j < 32; ++j) mx = fmaxf(mx, sS[head][lane][j]);
    float ev[32];
    float sum = 0.f;
#pragma unroll 8
    for (int j = 0; j < 32; ++j) { float e = __expf(sS[head][lane][j] - mx); ev[j] = e; sum += e; }
    float inv = 1.0f / sum;
#pragma unroll 8
    for (int j = 0; j < 32; ++j) sP[head][lane][j] = (_Float16)(ev[j] * inv);
  }
  __syncthreads();

  // O = P @ V  (M=32, N=16, K=32)
#pragma unroll
  for (int mt = 0; mt < 2; ++mt) {
    v16h a = load_a_frag((const _Float16*)&sP[head][mt * 16][0], 32, lane);
    H16 rb;
    {
      int n = lane & 15;
      int kb = (lane >> 4) * 16;
#pragma unroll
      for (int vv = 0; vv < 8; ++vv) {
        rb.h[2 * vv]     = vp[(size_t)(kb + 2 * vv) * 384 + n];
        rb.h[2 * vv + 1] = vp[(size_t)(kb + 2 * vv + 1) * 384 + n];
      }
    }
    v8f c = {};
    c = __builtin_amdgcn_wmma_f32_16x16x32_f16(false, a, false, rb.v, (short)0, c, false, false);
    int n = lane & 15;
#pragma unroll
    for (int r = 0; r < 8; ++r) {
      int m = mt * 16 + (lane >> 4) * 8 + r;
      obuf[(R + m) * Dn + head * HDn + n] = (_Float16)c[r];
    }
  }
}

// ---------------- residual add + LayerNorm (one wave per row of 128) ----------------
__global__ void add_ln_kernel(float* __restrict__ X, _Float16* __restrict__ Xh,
                              const float* __restrict__ addend,
                              const float* __restrict__ g, const float* __restrict__ bta) {
  const int lane = threadIdx.x & 31;
  const size_t row = (size_t)blockIdx.x * 8 + (threadIdx.x >> 5);
  float e[4];
  float s = 0.f, s2 = 0.f;
#pragma unroll
  for (int j = 0; j < 4; ++j) {
    int c = 32 * j + lane;
    float x = X[row * Dn + c] + addend[row * Dn + c];
    e[j] = x; s += x; s2 += x * x;
  }
#pragma unroll
  for (int m = 16; m >= 1; m >>= 1) {
    s  += __shfl_xor(s,  m, 32);
    s2 += __shfl_xor(s2, m, 32);
  }
  float mean = s * (1.0f / Dn);
  float var  = s2 * (1.0f / Dn) - mean * mean;
  float rstd = rsqrtf(var + EPSc);
#pragma unroll
  for (int j = 0; j < 4; ++j) {
    int c = 32 * j + lane;
    float y = (e[j] - mean) * rstd * g[c] + bta[c];
    X[row * Dn + c]  = y;
    Xh[row * Dn + c] = (_Float16)y;
  }
}

// ---------------- max-pool over the 32 neighbors ----------------
__global__ void maxpool_kernel(const float* __restrict__ X, _Float16* __restrict__ pooled) {
  const int bs = blockIdx.x;     // b*Sn+s
  const int t  = threadIdx.x;    // dim 0..127
  float m = -1e30f;
#pragma unroll 8
  for (int k = 0; k < Kn; ++k) m = fmaxf(m, X[((size_t)bs * Kn + k) * Dn + t]);
  pooled[(size_t)bs * Dn + t] = (_Float16)m;
}

// ---------------- outputs: transposed new_xyz + linear upsample ----------------
__global__ void newxyz_out_kernel(const float* __restrict__ nx, float* __restrict__ out) {
  int i = blockIdx.x * blockDim.x + threadIdx.x;
  if (i >= Bn * 3 * Sn) return;
  int b = i / (3 * Sn);
  int r = i % (3 * Sn);
  int c = r / Sn;
  int s = r % Sn;
  out[i] = nx[((size_t)b * Sn + s) * 3 + c];
}

__global__ void interp_kernel(const float* __restrict__ pooledOut, float* __restrict__ out) {
  size_t i = (size_t)blockIdx.x * blockDim.x + threadIdx.x;  // over B*DOUT*Nn
  int n = (int)(i % Nn);
  int c = (int)((i / Nn) % DOUT);
  int b = (int)(i / ((size_t)Nn * DOUT));
  float pos = n * ((float)(Sn - 1) / (float)(Nn - 1));
  int i0 = (int)floorf(pos);
  int i1 = min(i0 + 1, Sn - 1);
  float w = pos - (float)i0;
  float v0 = pooledOut[((size_t)b * Sn + i0) * DOUT + c];
  float v1 = pooledOut[((size_t)b * Sn + i1) * DOUT + c];
  out[(size_t)Bn * 3 * Sn + i] = v0 * (1.0f - w) + v1 * w;
}

// ---------------- host-side orchestration ----------------
extern "C" void kernel_launch(void* const* d_in, const int* in_sizes, int n_in,
                              void* d_out, int out_size, void* d_ws, size_t ws_size,
                              hipStream_t stream) {
  const float* xyz   = (const float*)d_in[0];
  const float* feat  = (const float*)d_in[1];
  const float* pe_w1 = (const float*)d_in[2];
  const float* pe_b1 = (const float*)d_in[3];
  const float* bn_g  = (const float*)d_in[4];
  const float* bn_b  = (const float*)d_in[5];
  const float* bn_m  = (const float*)d_in[6];
  const float* bn_v  = (const float*)d_in[7];
  const float* pe_w2 = (const float*)d_in[8];
  const float* pe_b2 = (const float*)d_in[9];
  const float* wqkv  = (const float*)d_in[10];
  const float* bqkv  = (const float*)d_in[11];
  const float* wo    = (const float*)d_in[12];
  const float* bo    = (const float*)d_in[13];
  const float* ln1g  = (const float*)d_in[14];
  const float* ln1b  = (const float*)d_in[15];
  const float* w1    = (const float*)d_in[16];
  const float* bf1   = (const float*)d_in[17];
  const float* w2    = (const float*)d_in[18];
  const float* bf2   = (const float*)d_in[19];
  const float* ln2g  = (const float*)d_in[20];
  const float* ln2b  = (const float*)d_in[21];
  const float* fcw   = (const float*)d_in[22];
  const float* fcb   = (const float*)d_in[23];
  float* out = (float*)d_out;

  // ---- workspace carve-out ----
  char* ws = (char*)d_ws;
  size_t off = 0;
  auto carve = [&](size_t bytes) -> void* {
    void* p = ws + off;
    off = (off + bytes + 255) & ~(size_t)255;
    return p;
  };
  float*     newXyz    = (float*)    carve((size_t)Bn * Sn * 3 * 4);
  int*       gidx      = (int*)      carve((size_t)Bn * Sn * Kn * 4);
  float*     X         = (float*)    carve((size_t)MT * Dn * 4);
  _Float16*  Xh        = (_Float16*) carve((size_t)MT * Dn * 2);
  _Float16*  QKV       = (_Float16*) carve((size_t)MT * 3 * Dn * 2);
  _Float16*  Obuf      = (_Float16*) carve((size_t)MT * Dn * 2);
  float*     tmpF      = (float*)    carve((size_t)MT * Dn * 4);
  _Float16*  Hbuf      = (_Float16*) carve((size_t)MT * DFFn * 2);
  _Float16*  pooled    = (_Float16*) carve((size_t)Bn * Sn * Dn * 2);
  float*     pooledOut = (float*)    carve((size_t)Bn * Sn * DOUT * 4);
  _Float16*  wqkvh     = (_Float16*) carve((size_t)NL * 3 * Dn * Dn * 2);
  _Float16*  woh       = (_Float16*) carve((size_t)NL * Dn * Dn * 2);
  _Float16*  w1h       = (_Float16*) carve((size_t)NL * DFFn * Dn * 2);
  _Float16*  w2h       = (_Float16*) carve((size_t)NL * Dn * DFFn * 2);
  _Float16*  fcwh      = (_Float16*) carve((size_t)DOUT * Dn * 2);

  // ---- weight conversion to f16 ----
  auto cvt = [&](const float* s, _Float16* d, int n) {
    cvt_f16_kernel<<<(n + 255) / 256, 256, 0, stream>>>(s, d, n);
  };
  cvt(wqkv, wqkvh, NL * 3 * Dn * Dn);
  cvt(wo,   woh,   NL * Dn * Dn);
  cvt(w1,   w1h,   NL * DFFn * Dn);
  cvt(w2,   w2h,   NL * Dn * DFFn);
  cvt(fcw,  fcwh,  DOUT * Dn);

  // ---- geometry: FPS, kNN, gather+PE ----
  fps_kernel<<<Bn, 256, 0, stream>>>(xyz, newXyz);
  knn_kernel<<<Bn * Sn, 256, 0, stream>>>(xyz, newXyz, gidx);
  build_x_kernel<<<Bn * Sn, Dn, 0, stream>>>(xyz, feat, gidx, pe_w1, pe_b1,
                                             bn_g, bn_b, bn_m, bn_v, pe_w2, pe_b2, X, Xh);

  // ---- transformer layers ----
  const int Mtiles = MT / 16;   // 8192 -> 1024 M-blocks of 8 tiles
  auto gblocks = [](int mtiles, int ntiles) { return (mtiles / 8) * (ntiles / 4); };
  for (int l = 0; l < NL; ++l) {
    // QKV projection: [MT,128] @ [128,384]^T -> f16
    wmma_gemm<Dn, false><<<gblocks(Mtiles, 24), 256, 0, stream>>>(
        Xh, wqkvh + (size_t)l * 3 * Dn * Dn, bqkv + (size_t)l * 3 * Dn, 24, QKV, nullptr);
    // fused attention per sequence
    attn_kernel<<<Bn * Sn, 256, 0, stream>>>(QKV, Obuf);
    // output projection -> fp32
    wmma_gemm<Dn, false><<<gblocks(Mtiles, 8), 256, 0, stream>>>(
        Obuf, woh + (size_t)l * Dn * Dn, bo + (size_t)l * Dn, 8, nullptr, tmpF);
    // x = LN(x + o)
    add_ln_kernel<<<MT / 8, 256, 0, stream>>>(X, Xh, tmpF, ln1g + (size_t)l * Dn, ln1b + (size_t)l * Dn);
    // FFN up + relu -> f16
    wmma_gemm<Dn, true><<<gblocks(Mtiles, 16), 256, 0, stream>>>(
        Xh, w1h + (size_t)l * DFFn * Dn, bf1 + (size_t)l * DFFn, 16, Hbuf, nullptr);
    // FFN down -> fp32
    wmma_gemm<DFFn, false><<<gblocks(Mtiles, 8), 256, 0, stream>>>(
        Hbuf, w2h + (size_t)l * Dn * DFFn, bf2 + (size_t)l * Dn, 8, nullptr, tmpF);
    // x = LN(x + ff)
    add_ln_kernel<<<MT / 8, 256, 0, stream>>>(X, Xh, tmpF, ln2g + (size_t)l * Dn, ln2b + (size_t)l * Dn);
  }

  // ---- maxpool over neighbors, FC, outputs ----
  maxpool_kernel<<<Bn * Sn, Dn, 0, stream>>>(X, pooled);
  wmma_gemm<Dn, false><<<gblocks(Bn * Sn / 16, 16), 256, 0, stream>>>(
      pooled, fcwh, fcb, 16, nullptr, pooledOut);
  newxyz_out_kernel<<<(Bn * 3 * Sn + 255) / 256, 256, 0, stream>>>(newXyz, out);
  interp_kernel<<<(int)(((size_t)Bn * DOUT * Nn) / 256), 256, 0, stream>>>(pooledOut, out);
}
